// StackGatedRNN_27719718928615
// MI455X (gfx1250) — matrix operand order repaired and checked by reading, not compiled, not run on hardware
//
#include <hip/hip_runtime.h>

// Problem constants (match reference)
#define BB   256
#define TT   512
#define DIN  64
#define HID  512

// LDS pitches (elements) chosen for bank-conflict-free 16B fragment chunks
#define XP   72    // xt tile pitch (bf16)   : 144B row stride
#define HP   520   // h1/h2 tile pitch (bf16): 1040B row stride
#define FP   528   // f32 state pitch        : 2112B row stride

typedef __attribute__((ext_vector_type(16))) __bf16 v16bf;
typedef __attribute__((ext_vector_type(8)))  float  v8f;
typedef __attribute__((ext_vector_type(4)))  float  f4;

union FragU { v16bf v; f4 f[2]; };

__device__ __forceinline__ unsigned short f2bf(float x) {
  unsigned int u = __float_as_uint(x);
  unsigned int r = u + 0x7FFFu + ((u >> 16) & 1u);   // round-to-nearest-even
  return (unsigned short)(r >> 16);
}

// gfx1250 has a hardware V_TANH_F32; use it when the builtin exists,
// otherwise a saturating exp-based tanh (v_exp_f32 is a single trans op).
__device__ __forceinline__ float fast_tanh(float x) {
#if __has_builtin(__builtin_amdgcn_tanhf)
  return __builtin_amdgcn_tanhf(x);
#else
  float e = __expf(2.0f * x);
  return 1.0f - __fdividef(2.0f, e + 1.0f);
#endif
}

__device__ __forceinline__ float fast_sigmoid(float x) {
  return __fdividef(1.0f, 1.0f + __expf(-x));
}

// A fragment: 16x32 (MxK) bf16. Lane l (l<16): row=l, K = 0..7 then 16..23.
// Lane l>=16: row=l-16, K = 8..15 then 24..31.  base -> element [0][k0].
__device__ __forceinline__ v16bf load_fragA(const unsigned short* base, int pitch, int lane) {
  const int row = lane & 15;
  const int hi  = lane >> 4;
  const unsigned short* p = base + row * pitch + hi * 8;
  FragU u;
  u.f[0] = *(const f4*)(p);
  u.f[1] = *(const f4*)(p + 16);
  return u.v;
}

// B fragment: 32x16 (KxN) bf16 with B[k][n] = W[n][k], W row-major [N][K].
// Lane l (l<16): col n=l, K = 0..7 then 16..23. Lane l>=16: K = 8..15 then 24..31.
// base -> element W[n0][k0].
__device__ __forceinline__ v16bf load_fragB(const unsigned short* base, int pitch, int lane) {
  const int n  = lane & 15;
  const int hi = lane >> 4;
  const unsigned short* p = base + n * pitch + hi * 8;
  FragU u;
  u.f[0] = *(const f4*)(p);
  u.f[1] = *(const f4*)(p + 16);
  return u.v;
}

// WGP-scope prefetch (scope field omitted -> 0 = WGP: fill all cache levels)
// of this lane's weight row, two 256B-line groups (512B) ahead of the stream.
// Issued once per 256B of stream advance (kk==0 copies only).
__device__ __forceinline__ void pf_row(const unsigned short* base, int pitch, int lane) {
  const unsigned short* p = base + (lane & 15) * pitch + (lane >> 4) * 16;
  asm volatile("global_prefetch_b8 %0, off offset:512" :: "v"((const void*)p));
}

__device__ __forceinline__ v8f wmma_bf16(v16bf a, v16bf b, v8f c) {
  return __builtin_amdgcn_wmma_f32_16x16x32_bf16(false, a, false, b, (short)0, c, false, false);
}

__global__ void f32_to_bf16_kernel(const float* __restrict__ src,
                                   unsigned short* __restrict__ dst, int n) {
  int i = blockIdx.x * blockDim.x + threadIdx.x;
  if (i < n) dst[i] = f2bf(src[i]);
}

// Persistent RNN kernel: 16 blocks, each owns 16 batch rows for all 512 steps.
// 8 waves/block: each wave owns 4 N-tiles (64 cols) of the H=512 GEMMs;
// waves 0..3 additionally own the 4 N-tiles of the D=64 output GEMM.
__global__ __launch_bounds__(256)
void StackGatedRNN_persistent_kernel(
    const float* __restrict__ x,
    const unsigned short* __restrict__ w1,  // l1_fc1_w [512][64]   bf16
    const unsigned short* __restrict__ w2,  // l1_fc2_w [512][512]
    const unsigned short* __restrict__ w3,  // l2_fc1_w [512][512]
    const unsigned short* __restrict__ w4,  // l2_fc2_w [512][512]
    const unsigned short* __restrict__ w5,  // l1_h2o_w [64][512]
    const unsigned short* __restrict__ w6,  // l2_h2o_w [64][512]
    const unsigned short* __restrict__ w7,  // h2g_w    [512][1024]
    const float* __restrict__ b1, const float* __restrict__ b2,
    const float* __restrict__ b3, const float* __restrict__ b4,
    const float* __restrict__ b5, const float* __restrict__ b6,
    const float* __restrict__ b7,
    float* __restrict__ out)
{
  __shared__ __align__(16) unsigned short sh_x [16 * XP];  // xt tile, bf16
  __shared__ __align__(16) unsigned short sh_h1[16 * HP];  // h1, bf16
  __shared__ __align__(16) unsigned short sh_h2[16 * HP];  // h2, bf16
  __shared__ __align__(16) float          sh_h2f[16 * FP]; // h2 master, f32
  __shared__ __align__(16) float          sh_u  [16 * FP]; // gate, f32

  const int tid     = threadIdx.x;
  const int wave    = tid >> 5;
  const int lane    = tid & 31;
  const int col     = lane & 15;
  const int hi      = lane >> 4;
  const int rowBase = blockIdx.x * 16;   // batch-row tile
  const int nbase   = wave * 64;         // this wave's first hidden column

  // init carried state
  for (int i = tid; i < 16 * HP; i += blockDim.x) { sh_h1[i] = 0; sh_h2[i] = 0; }
  for (int i = tid; i < 16 * FP; i += blockDim.x) { sh_h2f[i] = 0.0f; sh_u[i] = 1.0f; }

  // biases are time-invariant per (wave, lane): hoist out of the T loop
  float bias1[4], bias2[4], bias3[4];
  #pragma unroll
  for (int j = 0; j < 4; ++j) {
    int n0 = nbase + j * 16;
    bias1[j] = b1[n0 + col] + b2[n0 + col];
    bias2[j] = b3[n0 + col] + b4[n0 + col];
    bias3[j] = b7[n0 + col];
  }
  float biasO = 0.0f;
  if (wave < 4) biasO = b5[wave * 16 + col] + b6[wave * 16 + col];

  __syncthreads();

  #pragma unroll 1
  for (int t = 0; t < TT; ++t) {
    // ---- stage X: load x[:,t,:] tile (16x64 f32) -> bf16 LDS -------------
    {
      int r = tid >> 4;            // 0..15 (batch row in tile)
      int c = (tid & 15) * 4;      // 0..60
      f4 v = *(const f4*)(x + ((size_t)(rowBase + r) * TT + t) * DIN + c);
      sh_x[r * XP + c + 0] = f2bf(v.x);
      sh_x[r * XP + c + 1] = f2bf(v.y);
      sh_x[r * XP + c + 2] = f2bf(v.z);
      sh_x[r * XP + c + 3] = f2bf(v.w);
    }
    __syncthreads();

    // ---- GEMM1: h1_new = tanh(xt@W1^T + h1@W2^T + b) ---------------------
    v8f acc[4];
    #pragma unroll
    for (int j = 0; j < 4; ++j) {
      v8f a;
      #pragma unroll
      for (int r = 0; r < 8; ++r) a[r] = bias1[j];
      acc[j] = a;
    }
    #pragma unroll
    for (int k0 = 0; k0 < DIN; k0 += 32) {
      v16bf af = load_fragA(sh_x + k0, XP, lane);
      #pragma unroll
      for (int j = 0; j < 4; ++j) {
        const unsigned short* wb = w1 + (size_t)(nbase + j * 16) * DIN + k0;
        acc[j] = wmma_bf16(af, load_fragB(wb, DIN, lane), acc[j]);
      }
    }
    #pragma unroll 1
    for (int kb = 0; kb < HID; kb += 128) {
      #pragma unroll
      for (int kk = 0; kk < 128; kk += 32) {
        const int k0 = kb + kk;
        v16bf af = load_fragA(sh_h1 + k0, HP, lane);
        #pragma unroll
        for (int j = 0; j < 4; ++j) {
          const unsigned short* wb = w2 + (size_t)(nbase + j * 16) * HID + k0;
          if (kk == 0) pf_row(wb, HID, lane);
          acc[j] = wmma_bf16(af, load_fragB(wb, HID, lane), acc[j]);
        }
      }
    }
    __syncthreads();   // all reads of old h1 complete
    #pragma unroll
    for (int j = 0; j < 4; ++j) {
      int n0 = nbase + j * 16;
      #pragma unroll
      for (int r = 0; r < 8; ++r) {
        int row = r + hi * 8;   // C/D layout: VGPR r -> M = r + 8*(lane>=16)
        sh_h1[row * HP + n0 + col] = f2bf(fast_tanh(acc[j][r]));
      }
    }
    __syncthreads();

    // ---- GEMM2: h2_new = tanh(h1@W3^T + h2@W4^T + b) ---------------------
    #pragma unroll
    for (int j = 0; j < 4; ++j) {
      v8f a;
      #pragma unroll
      for (int r = 0; r < 8; ++r) a[r] = bias2[j];
      acc[j] = a;
    }
    #pragma unroll 1
    for (int kb = 0; kb < HID; kb += 128) {
      #pragma unroll
      for (int kk = 0; kk < 128; kk += 32) {
        const int k0 = kb + kk;
        v16bf a1 = load_fragA(sh_h1 + k0, HP, lane);
        v16bf a2 = load_fragA(sh_h2 + k0, HP, lane);
        #pragma unroll
        for (int j = 0; j < 4; ++j) {
          const unsigned short* wb3 = w3 + (size_t)(nbase + j * 16) * HID + k0;
          const unsigned short* wb4 = w4 + (size_t)(nbase + j * 16) * HID + k0;
          if (kk == 0) { pf_row(wb3, HID, lane); pf_row(wb4, HID, lane); }
          acc[j] = wmma_bf16(a1, load_fragB(wb3, HID, lane), acc[j]);
          acc[j] = wmma_bf16(a2, load_fragB(wb4, HID, lane), acc[j]);
        }
      }
    }
    __syncthreads();   // all reads of old h2 complete

    // ---- gate: h2 = u*tanh(acc) + (1-u)*h2  (f32 master + bf16 copy) -----
    #pragma unroll
    for (int j = 0; j < 4; ++j) {
      int n0 = nbase + j * 16;
      #pragma unroll
      for (int r = 0; r < 8; ++r) {
        int row = r + hi * 8;
        int fi  = row * FP + n0 + col;
        float h2new = fast_tanh(acc[j][r]);
        float uu    = sh_u[fi];
        float h2c   = uu * h2new + (1.0f - uu) * sh_h2f[fi];
        sh_h2f[fi] = h2c;
        sh_h2[row * HP + n0 + col] = f2bf(h2c);
      }
    }
    __syncthreads();

    // ---- out GEMM (waves 0..3): out = h1@W5^T + h2@W6^T + b --------------
    if (wave < 4) {
      v8f oacc;
      #pragma unroll
      for (int r = 0; r < 8; ++r) oacc[r] = biasO;
      #pragma unroll 1
      for (int kb = 0; kb < HID; kb += 128) {
        #pragma unroll
        for (int kk = 0; kk < 128; kk += 32) {
          const int k0 = kb + kk;
          const unsigned short* wb5 = w5 + (size_t)(wave * 16) * HID + k0;
          const unsigned short* wb6 = w6 + (size_t)(wave * 16) * HID + k0;
          if (kk == 0) { pf_row(wb5, HID, lane); pf_row(wb6, HID, lane); }
          v16bf a1 = load_fragA(sh_h1 + k0, HP, lane);
          v16bf a2 = load_fragA(sh_h2 + k0, HP, lane);
          oacc = wmma_bf16(a1, load_fragB(wb5, HID, lane), oacc);
          oacc = wmma_bf16(a2, load_fragB(wb6, HID, lane), oacc);
        }
      }
      #pragma unroll
      for (int r = 0; r < 8; ++r) {
        int row = r + hi * 8;
        out[((size_t)(rowBase + row) * TT + t) * DIN + wave * 16 + col] = oacc[r];
      }
    }

    // ---- gate GEMM: u = round(sigmoid([h1,h2]@W7^T + b)) -----------------
    v8f uacc[4];
    #pragma unroll
    for (int j = 0; j < 4; ++j) {
      v8f a;
      #pragma unroll
      for (int r = 0; r < 8; ++r) a[r] = bias3[j];
      uacc[j] = a;
    }
    #pragma unroll 1
    for (int kb = 0; kb < HID; kb += 128) {
      #pragma unroll
      for (int kk = 0; kk < 128; kk += 32) {
        const int k0 = kb + kk;
        v16bf a1 = load_fragA(sh_h1 + k0, HP, lane);
        v16bf a2 = load_fragA(sh_h2 + k0, HP, lane);
        #pragma unroll
        for (int j = 0; j < 4; ++j) {
          const unsigned short* wb = w7 + (size_t)(nbase + j * 16) * (2 * HID);
          if (kk == 0) { pf_row(wb + k0, 2 * HID, lane); pf_row(wb + HID + k0, 2 * HID, lane); }
          uacc[j] = wmma_bf16(a1, load_fragB(wb + k0,       2 * HID, lane), uacc[j]);
          uacc[j] = wmma_bf16(a2, load_fragB(wb + HID + k0, 2 * HID, lane), uacc[j]);
        }
      }
    }
    // sh_u readers (gate stage) already passed a barrier; safe to overwrite
    #pragma unroll
    for (int j = 0; j < 4; ++j) {
      int n0 = nbase + j * 16;
      #pragma unroll
      for (int r = 0; r < 8; ++r) {
        int row = r + hi * 8;
        sh_u[row * FP + n0 + col] = rintf(fast_sigmoid(uacc[j][r]));
      }
    }
    __syncthreads();   // end of timestep
  }
}

extern "C" void kernel_launch(void* const* d_in, const int* in_sizes, int n_in,
                              void* d_out, int out_size, void* d_ws, size_t ws_size,
                              hipStream_t stream) {
  (void)in_sizes; (void)n_in; (void)out_size; (void)ws_size;

  const float* x        = (const float*)d_in[0];
  const float* l1_fc1_w = (const float*)d_in[1];
  const float* l1_fc1_b = (const float*)d_in[2];
  const float* l1_fc2_w = (const float*)d_in[3];
  const float* l1_fc2_b = (const float*)d_in[4];
  const float* l2_fc1_w = (const float*)d_in[5];
  const float* l2_fc1_b = (const float*)d_in[6];
  const float* l2_fc2_w = (const float*)d_in[7];
  const float* l2_fc2_b = (const float*)d_in[8];
  const float* l1_h2o_w = (const float*)d_in[9];
  const float* l1_h2o_b = (const float*)d_in[10];
  const float* l2_h2o_w = (const float*)d_in[11];
  const float* l2_h2o_b = (const float*)d_in[12];
  const float* h2g_w    = (const float*)d_in[13];
  const float* h2g_b    = (const float*)d_in[14];

  // bf16 weight cache in workspace (1,409,024 elems = 2.82 MB; L2-resident)
  unsigned short* ws = (unsigned short*)d_ws;
  unsigned short* w1 = ws;                       // 512*64
  unsigned short* w2 = w1 + 512 * 64;            // 512*512
  unsigned short* w3 = w2 + 512 * 512;
  unsigned short* w4 = w3 + 512 * 512;
  unsigned short* w5 = w4 + 512 * 512;           // 64*512
  unsigned short* w6 = w5 + 64 * 512;
  unsigned short* w7 = w6 + 64 * 512;            // 512*1024

  auto cvt = [&](const float* s, unsigned short* d, int n) {
    f32_to_bf16_kernel<<<(n + 255) / 256, 256, 0, stream>>>(s, d, n);
  };
  cvt(l1_fc1_w, w1, 512 * 64);
  cvt(l1_fc2_w, w2, 512 * 512);
  cvt(l2_fc1_w, w3, 512 * 512);
  cvt(l2_fc2_w, w4, 512 * 512);
  cvt(l1_h2o_w, w5, 64 * 512);
  cvt(l2_h2o_w, w6, 64 * 512);
  cvt(h2g_w,    w7, 512 * 1024);

  StackGatedRNN_persistent_kernel<<<dim3(BB / 16), dim3(256), 0, stream>>>(
      x, w1, w2, w3, w4, w5, w6, w7,
      l1_fc1_b, l1_fc2_b, l2_fc1_b, l2_fc2_b, l1_h2o_b, l2_h2o_b, h2g_b,
      (float*)d_out);
}
